// MultiHeadSelfAttention_74234214744417
// MI455X (gfx1250) — matrix operand order repaired and checked by reading, not compiled, not run on hardware
//
#include <hip/hip_runtime.h>
#include <hip/hip_bf16.h>

typedef __attribute__((ext_vector_type(16))) _Float16 v16h;
typedef __attribute__((ext_vector_type(8)))  _Float16 v8h;
typedef __attribute__((ext_vector_type(8)))  float    v8f;

#define D_MODEL 2048
#define N_HEADS 16
#define D_HEAD  128
#define SEQ_N   2048
#define BATCH   2
#define ROWS    (BATCH * SEQ_N)          // 4096
#define SOFTMAX_SCALE 0.08838834764831845f  // 1/sqrt(128)

// ---------------------------------------------------------------------------
// WMMA fragment loaders (16x16x32 f16, wave32).
// A (16xK=32): lane L -> row m = L&15, khalf = L>>4; elems 0..7 = K[khalf*8..],
//              elems 8..15 = K[16+khalf*8..]   (ISA 05_wmma 16-bit A layout)
// B (K=32x16): lane L -> col n = L&15, khalf = L>>4; elems 0..7 = K[khalf*8..], elems 8..15 = K[16+khalf*8..]
//              loaded from a row-major "B-transposed" buffer (col n contiguous)
// C/D (16x16): VGPR v, lane L -> row v + 8*(L>>4), col L&15
// ---------------------------------------------------------------------------
__device__ __forceinline__ v16h load_a_frag(const _Float16* base, int ld, int lane) {
    int m  = lane & 15;
    int kh = (lane >> 4) * 8;
    const _Float16* p = base + (size_t)m * ld;
    v8h lo = *(const v8h*)(p + kh);
    v8h hi = *(const v8h*)(p + 16 + kh);
    v16h r;
#pragma unroll
    for (int i = 0; i < 8; ++i) { r[i] = lo[i]; r[8 + i] = hi[i]; }
    return r;
}

__device__ __forceinline__ v16h load_b_frag(const _Float16* baseT, int ld, int lane) {
    int n  = lane & 15;
    int kh = (lane >> 4) * 8;
    const _Float16* p = baseT + (size_t)n * ld;
    v8h lo = *(const v8h*)(p + kh);
    v8h hi = *(const v8h*)(p + 16 + kh);
    v16h r;
#pragma unroll
    for (int i = 0; i < 8; ++i) { r[i] = lo[i]; r[8 + i] = hi[i]; }
    return r;
}

__device__ __forceinline__ v8f wmma_f16(v16h a, v16h b, v8f c) {
    return __builtin_amdgcn_wmma_f32_16x16x32_f16(false, a, false, b, (short)0, c, false, false);
}

// ---------------------------------------------------------------------------
// Kernel 1: fp32 -> f16 elementwise convert
// ---------------------------------------------------------------------------
__global__ void __launch_bounds__(256) cvt_f16(const float* __restrict__ src,
                                               _Float16* __restrict__ dst, int n) {
    int i = (blockIdx.x * blockDim.x + threadIdx.x) * 4;
    if (i + 3 < n) {
        float4 v = *(const float4*)(src + i);
        dst[i + 0] = (_Float16)v.x;
        dst[i + 1] = (_Float16)v.y;
        dst[i + 2] = (_Float16)v.z;
        dst[i + 3] = (_Float16)v.w;
    }
}

// ---------------------------------------------------------------------------
// Kernel 2: 32x32 LDS-tiled transpose + convert: dst[n*dim+k] = (f16)src[k*dim+n]
// ---------------------------------------------------------------------------
__global__ void __launch_bounds__(256) transpose_cvt(const float* __restrict__ src,
                                                     _Float16* __restrict__ dst, int dim) {
    __shared__ float tile[32][33];
    int tx = threadIdx.x, ty = threadIdx.y;
    int x = blockIdx.x * 32 + tx;          // src col
    int y = blockIdx.y * 32;               // src row base
#pragma unroll
    for (int j = 0; j < 4; ++j)
        tile[ty + j * 8][tx] = src[(size_t)(y + ty + j * 8) * dim + x];
    __syncthreads();
    int nx = blockIdx.y * 32 + tx;         // dst col (= src row)
    int ny = blockIdx.x * 32;              // dst row base (= src col)
#pragma unroll
    for (int j = 0; j < 4; ++j)
        dst[(size_t)(ny + ty + j * 8) * dim + nx] = (_Float16)tile[tx][ty + j * 8];
}

// ---------------------------------------------------------------------------
// Kernel 3/5: WMMA GEMM  C[M,Nn] = A[M,K] * B[K,Nn], B given as BT[Nn,K] f16.
// Block tile 128x128, 8 waves of 32x64, BK=32 staged in LDS.
// MODE 0: store fp32 C.   MODE 1: fused QKV epilogue with interleaved RoPE.
// ---------------------------------------------------------------------------
template <int MODE>
__global__ void __launch_bounds__(256, 1)
gemm_wmma(const _Float16* __restrict__ A, const _Float16* __restrict__ BT,
          int M, int Nn, int K,
          float* __restrict__ Cf,
          _Float16* __restrict__ Qh, _Float16* __restrict__ Kh,
          _Float16* __restrict__ Vh,
          const float* __restrict__ cosT, const float* __restrict__ sinT) {
    __shared__ _Float16 As[128 * 40];
    __shared__ _Float16 Bs[128 * 40];

    int lane = threadIdx.x & 31;
    int w    = threadIdx.x >> 5;     // 0..7
    int wm   = w & 3;                // row quadrant (32 rows)
    int wn   = w >> 2;               // col half (64 cols)
    int n0 = blockIdx.x * 128;
    int m0 = blockIdx.y * 128;

    v8f acc[2][4];
#pragma unroll
    for (int i = 0; i < 2; ++i)
#pragma unroll
        for (int j = 0; j < 4; ++j) acc[i][j] = {};

    int trow = threadIdx.x >> 1;     // 0..127
    int tseg = threadIdx.x & 1;      // 16 halfs each

    for (int kb = 0; kb < K; kb += 32) {
        const _Float16* as = A  + (size_t)(m0 + trow) * K + kb + tseg * 16;
        const _Float16* bs = BT + (size_t)(n0 + trow) * K + kb + tseg * 16;
        *(v16h*)(As + trow * 40 + tseg * 16) = *(const v16h*)as;
        *(v16h*)(Bs + trow * 40 + tseg * 16) = *(const v16h*)bs;
        if (kb + 32 < K) {  // hint next tiles into cache (global_prefetch_b8)
            __builtin_prefetch(as + 32, 0, 0);
            __builtin_prefetch(bs + 32, 0, 0);
        }
        __syncthreads();

        v16h af[2];
#pragma unroll
        for (int i = 0; i < 2; ++i)
            af[i] = load_a_frag(As + (wm * 32 + i * 16) * 40, 40, lane);
#pragma unroll
        for (int j = 0; j < 4; ++j) {
            v16h bf = load_b_frag(Bs + (wn * 64 + j * 16) * 40, 40, lane);
#pragma unroll
            for (int i = 0; i < 2; ++i) acc[i][j] = wmma_f16(af[i], bf, acc[i][j]);
        }
        __syncthreads();
    }

    // ---- epilogue ----
#pragma unroll
    for (int i = 0; i < 2; ++i) {
#pragma unroll
        for (int j = 0; j < 4; ++j) {
#pragma unroll
            for (int v = 0; v < 8; ++v) {
                int row = m0 + wm * 32 + i * 16 + v + ((lane >> 4) << 3);
                int col = n0 + wn * 64 + j * 16 + (lane & 15);
                float val = acc[i][j][v];
                if (MODE == 0) {
                    Cf[(size_t)row * Nn + col] = val;
                } else {
                    // cols [0,2048)=Q, [2048,4096)=K (RoPE'd), [4096,6144)=V
                    float partner = __shfl_xor(val, 1);   // uniform across wave
                    int c     = col & 2047;
                    int n_tok = row & (SEQ_N - 1);
                    if (col < 4096) {
                        int jj = (c & 127) >> 1;
                        float cs = cosT[n_tok * 64 + jj];
                        float sn = sinT[n_tok * 64 + jj];
                        float r = ((c & 1) == 0) ? (val * cs - partner * sn)
                                                 : (partner * sn + val * cs);
                        _Float16 hv = (_Float16)r;
                        if (col < 2048) Qh[(size_t)row * 2048 + c] = hv;
                        else            Kh[(size_t)row * 2048 + c] = hv;
                    } else {
                        Vh[(size_t)row * 2048 + c] = (_Float16)val;
                    }
                }
            }
        }
    }
}

// ---------------------------------------------------------------------------
// Kernel 4: causal flash attention. 4 waves/block, 16 query rows per wave,
// streaming 32-key chunks through LDS with online softmax.
// __launch_bounds__(128, 1): ~200 VGPRs/lane live (Q frags + 16x128 f32 acc +
// softmax state); allow the allocator to keep everything resident, no spills.
// ---------------------------------------------------------------------------
__global__ void __launch_bounds__(128, 1)
attn_kernel(const _Float16* __restrict__ Qh, const _Float16* __restrict__ Kh,
            const _Float16* __restrict__ Vh, _Float16* __restrict__ Oh) {
    __shared__ _Float16 Ks[32 * 136];      // K chunk, row-major [key][dh]
    __shared__ _Float16 Vt[128 * 40];      // V chunk, transposed [dh][key]
    __shared__ _Float16 Pt[4 * 16 * 40];   // per-wave P staging [16][32]

    int lane = threadIdx.x & 31;
    int w    = threadIdx.x >> 5;           // 0..3
    int idx  = blockIdx.x;
    int qt = idx & 31;                     // N/64 = 32 query tiles
    int bh = idx >> 5;
    int h  = bh & 15;
    int b  = bh >> 4;
    int q0    = qt * 64;
    int qbase = q0 + w * 16;
    size_t rowQ = (size_t)(b * SEQ_N + qbase);

    // Q fragments for this wave: 16 rows x 128 dh = 4 x (16x32) A-fragments
    v16h qf[4];
    const _Float16* qptr = Qh + rowQ * 2048 + h * 128;
#pragma unroll
    for (int kk = 0; kk < 4; ++kk) qf[kk] = load_a_frag(qptr + kk * 32, 2048, lane);

    float rowM[8], rowL[8];
    v8f   o[8];
#pragma unroll
    for (int v = 0; v < 8; ++v) { rowM[v] = -3e38f; rowL[v] = 0.f; }
#pragma unroll
    for (int d = 0; d < 8; ++d) o[d] = {};

    int nCh = q0 / 32 + 2;                 // uniform across the block (causal)
    for (int kc = 0; kc < nCh; ++kc) {
        int k0 = kc * 32;
        {   // cooperative K/V chunk load (128 threads, 32 keys x 128 dh each)
            int row  = threadIdx.x >> 2;   // key 0..31
            int part = threadIdx.x & 3;    // 32-dh segment
            const _Float16* ks = Kh + (size_t)(b * SEQ_N + k0 + row) * 2048 + h * 128 + part * 32;
            *(v16h*)(Ks + row * 136 + part * 32)      = *(const v16h*)ks;
            *(v16h*)(Ks + row * 136 + part * 32 + 16) = *(const v16h*)(ks + 16);
            const _Float16* vs = Vh + (size_t)(b * SEQ_N + k0 + row) * 2048 + h * 128 + part * 32;
            v16h va = *(const v16h*)vs;
            v16h vb = *(const v16h*)(vs + 16);
#pragma unroll
            for (int i = 0; i < 16; ++i) {
                Vt[(part * 32 + i) * 40 + row]        = va[i];
                Vt[(part * 32 + 16 + i) * 40 + row]   = vb[i];
            }
        }
        __syncthreads();

        // S = Q * K^T  (two 16x16 score tiles over the 32-key chunk)
        v8f s0 = {}, s1 = {};
#pragma unroll
        for (int kk = 0; kk < 4; ++kk) {
            v16h bk0 = load_b_frag(Ks + kk * 32, 136, lane);
            s0 = wmma_f16(qf[kk], bk0, s0);
        }
#pragma unroll
        for (int kk = 0; kk < 4; ++kk) {
            v16h bk1 = load_b_frag(Ks + 16 * 136 + kk * 32, 136, lane);
            s1 = wmma_f16(qf[kk], bk1, s1);
        }

        _Float16* Pw = Pt + w * 640;
        int keyc = k0 + (lane & 15);
#pragma unroll
        for (int v = 0; v < 8; ++v) {
            int qrow = qbase + v + ((lane >> 4) << 3);
            float a  = s0[v] * SOFTMAX_SCALE; if (keyc > qrow)      a = -3e38f;
            float bb = s1[v] * SOFTMAX_SCALE; if (keyc + 16 > qrow) bb = -3e38f;
            // row max across the 16 lanes holding this row's columns
            float cm = fmaxf(a, bb);
            cm = fmaxf(cm, __shfl_xor(cm, 1));
            cm = fmaxf(cm, __shfl_xor(cm, 2));
            cm = fmaxf(cm, __shfl_xor(cm, 4));
            cm = fmaxf(cm, __shfl_xor(cm, 8));
            float nm  = fmaxf(rowM[v], cm);
            float fac = __expf(rowM[v] - nm);
            rowM[v] = nm;
            float p0 = __expf(a - nm);
            float p1 = __expf(bb - nm);
            float rs = p0 + p1;
            rs += __shfl_xor(rs, 1);
            rs += __shfl_xor(rs, 2);
            rs += __shfl_xor(rs, 4);
            rs += __shfl_xor(rs, 8);
            rowL[v] = rowL[v] * fac + rs;
#pragma unroll
            for (int d = 0; d < 8; ++d) o[d][v] *= fac;
            int pr = (v + ((lane >> 4) << 3)) * 40;
            Pw[pr + (lane & 15)]      = (_Float16)p0;
            Pw[pr + 16 + (lane & 15)] = (_Float16)p1;
        }
        // wave-private LDS C-layout -> A-layout transpose; wait for ds stores
        asm volatile("s_wait_dscnt 0x0" ::: "memory");
        v16h ap = load_a_frag(Pw, 40, lane);
#pragma unroll
        for (int d = 0; d < 8; ++d) {
            v16h bv = load_b_frag(Vt + d * 16 * 40, 40, lane);
            o[d] = wmma_f16(ap, bv, o[d]);
        }
        __syncthreads();   // protect Ks/Vt before next chunk overwrite
    }

    // normalize and store attention output (f16, [token, 2048])
#pragma unroll
    for (int v = 0; v < 8; ++v) {
        float inv = 1.0f / rowL[v];
        size_t row = rowQ + v + ((lane >> 4) << 3);
#pragma unroll
        for (int d = 0; d < 8; ++d)
            Oh[row * 2048 + h * 128 + d * 16 + (lane & 15)] = (_Float16)(o[d][v] * inv);
    }
}

// ---------------------------------------------------------------------------
// Host launcher
// ---------------------------------------------------------------------------
extern "C" void kernel_launch(void* const* d_in, const int* in_sizes, int n_in,
                              void* d_out, int out_size, void* d_ws, size_t ws_size,
                              hipStream_t stream) {
    (void)in_sizes; (void)n_in; (void)out_size; (void)ws_size;
    const float* x    = (const float*)d_in[0];
    const float* wq   = (const float*)d_in[1];
    const float* wk   = (const float*)d_in[2];
    const float* wv   = (const float*)d_in[3];
    const float* wo   = (const float*)d_in[4];
    const float* cosT = (const float*)d_in[5];
    const float* sinT = (const float*)d_in[6];

    const size_t XD = (size_t)ROWS * D_MODEL;        // 8,388,608 elems
    const size_t WD = (size_t)D_MODEL * D_MODEL;     // 4,194,304 elems

    char* p = (char*)d_ws;
    _Float16* xh  = (_Float16*)p; p += XD * 2;       // x in f16
    _Float16* wT  = (_Float16*)p; p += 3 * WD * 2;   // [WqT|WkT|WvT] rows = out cols
    _Float16* woT = (_Float16*)p; p += WD * 2;
    _Float16* Qh  = (_Float16*)p; p += XD * 2;
    _Float16* Kh  = (_Float16*)p; p += XD * 2;
    _Float16* Vh  = (_Float16*)p; p += XD * 2;
    _Float16* Oh  = (_Float16*)p; p += XD * 2;

    // 1) convert activations
    cvt_f16<<<(int)(XD / (256 * 4)), 256, 0, stream>>>(x, xh, (int)XD);

    // 2) transpose-convert weights
    dim3 tb(32, 8), tg(D_MODEL / 32, D_MODEL / 32);
    transpose_cvt<<<tg, tb, 0, stream>>>(wq, wT,          D_MODEL);
    transpose_cvt<<<tg, tb, 0, stream>>>(wk, wT + WD,     D_MODEL);
    transpose_cvt<<<tg, tb, 0, stream>>>(wv, wT + 2 * WD, D_MODEL);
    transpose_cvt<<<tg, tb, 0, stream>>>(wo, woT,         D_MODEL);

    // 3) fused QKV projection + RoPE epilogue  (M=4096, N=6144, K=2048)
    gemm_wmma<1><<<dim3(3 * D_MODEL / 128, ROWS / 128), 256, 0, stream>>>(
        xh, wT, ROWS, 3 * D_MODEL, D_MODEL,
        (float*)nullptr, Qh, Kh, Vh, cosT, sinT);

    // 4) causal flash attention
    attn_kernel<<<BATCH * N_HEADS * (SEQ_N / 64), 128, 0, stream>>>(Qh, Kh, Vh, Oh);

    // 5) output projection -> fp32 d_out  (M=4096, N=2048, K=2048)
    gemm_wmma<0><<<dim3(D_MODEL / 128, ROWS / 128), 256, 0, stream>>>(
        Oh, woT, ROWS, D_MODEL, D_MODEL,
        (float*)d_out, (_Float16*)nullptr, (_Float16*)nullptr, (_Float16*)nullptr,
        (const float*)nullptr, (const float*)nullptr);
}
